// SpikingNetwork_56006373539872
// MI455X (gfx1250) — matrix operand order repaired and checked by reading, not compiled, not run on hardware
//
#include <hip/hip_runtime.h>
#include <hip/hip_bf16.h>

typedef __attribute__((ext_vector_type(16))) _Float16 v16h;
typedef __attribute__((ext_vector_type(8)))  _Float16 v8h;
typedef __attribute__((ext_vector_type(8)))  float    v8f;

#define T_STEPS  20
#define DECAY_F  0.9f
#define B_TOT    16384
#define IN_DIM   784
#define KP1      800      // layer-1 K padded to multiple of 32 (25 K-steps)
#define H_DIM    1024
#define OUT_DIM  10
#define OUT_PAD  16

#define NWAVE    16       // 512 threads
#define NT       4        // 64 hidden columns per wave (4 16-wide tiles)

// ---------- stateless per-(element,timestep) RNG (deterministic) ----------
__device__ __forceinline__ uint32_t pcg_hash(uint32_t x) {
  x = x * 747796405u + 2891336453u;
  uint32_t w = ((x >> ((x >> 28u) + 4u)) ^ x) * 277803737u;
  return (w >> 22u) ^ w;
}
__device__ __forceinline__ float rand_u01(uint32_t idx, uint32_t t) {
  uint32_t h = pcg_hash(idx ^ (t * 0x9E3779B9u + 0x85EBCA6Bu));
  return (float)(h >> 8) * (1.0f / 16777216.0f);
}

__device__ __forceinline__ v16h cat16(v8h lo, v8h hi) {
  return __builtin_shufflevector(lo, hi, 0,1,2,3,4,5,6,7,8,9,10,11,12,13,14,15);
}

// A fragment (16xK): lane = row M, lane-group g holds K runs [g*8..g*8+7], [16+g*8..]
__device__ __forceinline__ v16h ldA(const _Float16* p) {        // p = base + lx*stride + lg*8 + kb
  return cat16(*(const v8h*)p, *(const v8h*)(p + 16));
}
// B fragment (Kx16): lane = col N, lane-group g holds K runs [g*8..g*8+7], [16+g*8..]
__device__ __forceinline__ v16h ldB(const _Float16* p) {        // p = row + lg*8 + kb
  return cat16(*(const v8h*)p, *(const v8h*)(p + 16));
}
__device__ __forceinline__ v8f wmma16(v16h a, v16h b, v8f c) {
  return __builtin_amdgcn_wmma_f32_16x16x32_f16(false, a, false, b, (short)0, c, false, false);
}

// ---------- weight conversion prologues ----------
__global__ void conv_w1(const float* __restrict__ W1, _Float16* __restrict__ W1h) {
  int i = blockIdx.x * 256 + threadIdx.x;
  if (i >= H_DIM * KP1) return;
  int n = i / KP1, k = i - n * KP1;
  W1h[i] = (k < IN_DIM) ? (_Float16)W1[(size_t)n * IN_DIM + k] : (_Float16)0.0f;
}
__global__ void conv_w2(const float* __restrict__ W2, _Float16* __restrict__ W2h) {
  int i = blockIdx.x * 256 + threadIdx.x;
  if (i >= OUT_PAD * H_DIM) return;
  int n = i / H_DIM, k = i - n * H_DIM;
  W2h[i] = (n < OUT_DIM) ? (_Float16)W2[(size_t)n * H_DIM + k] : (_Float16)0.0f;
}

// ---------- fused SNN: all 20 timesteps, membrane state in registers ----------
__global__ __launch_bounds__(NWAVE * 32) void snn_fused(
    const float* __restrict__ input, const float* __restrict__ b1,
    const float* __restrict__ b2, const _Float16* __restrict__ W1h,
    const _Float16* __restrict__ W2h, float* __restrict__ counts,
    float* __restrict__ trains)
{
  __shared__ __align__(16) unsigned char smem[16 * KP1 * 2 + 16 * H_DIM * 2];
  _Float16* lds_spk = (_Float16*)smem;                       // 25600 B (layer-1 A)
  _Float16* lds_s1  = (_Float16*)(smem + 16 * KP1 * 2);      // 32768 B (layer-2 A)
  float*    lds_red = (float*)smem;                          // 16384 B, aliases lds_spk

  const int tid  = threadIdx.x;
  const int lane = tid & 31;
  const int wave = tid >> 5;       // 0..15
  const int lx   = lane & 15;
  const int lg   = lane >> 4;
  const int rowBase = blockIdx.x * 16;

  v8f V1[NT];                      // hidden membrane: wave owns 64 columns
  #pragma unroll
  for (int nt = 0; nt < NT; ++nt) { v8f z = {}; V1[nt] = z; }
  v8f V2  = {};                    // output membrane (meaningful on wave 0)
  v8f cnt = {};

  float bias1[NT];
  #pragma unroll
  for (int nt = 0; nt < NT; ++nt) bias1[nt] = b1[wave * 64 + nt * 16 + lx];
  const float bias2 = (lx < OUT_DIM) ? b2[lx] : 0.0f;

  // loop-invariant fragment base pointers
  const _Float16* w1p[NT];
  #pragma unroll
  for (int nt = 0; nt < NT; ++nt)
    w1p[nt] = W1h + (size_t)(wave * 64 + nt * 16 + lx) * KP1 + lg * 8;
  const _Float16* apk = lds_spk + lx * KP1 + lg * 8;          // layer-1 A base
  const _Float16* a2p = lds_s1 + lx * H_DIM + lg * 8;         // layer-2 A base
  const _Float16* w2p = W2h + (size_t)lx * H_DIM + lg * 8;    // layer-2 B base

  for (int t = 0; t < T_STEPS; ++t) {
    // ---- Poisson rate-coded input spikes -> LDS f16 (zero-padded K) ----
    for (int idx = tid; idx < 16 * KP1; idx += NWAVE * 32) {
      int r = idx / KP1, c = idx - r * KP1;
      _Float16 s = (_Float16)0.0f;
      if (c < IN_DIM) {
        uint32_t gi = (uint32_t)((rowBase + r) * IN_DIM + c);
        float p = input[(size_t)(rowBase + r) * IN_DIM + c] * (1.0f / T_STEPS);
        s = (rand_u01(gi, (uint32_t)t) < p) ? (_Float16)1.0f : (_Float16)0.0f;
      }
      lds_spk[idx] = s;
    }
    __syncthreads();

    // ---- layer 1: fold decay + bias into the WMMA accumulator ----
    #pragma unroll
    for (int nt = 0; nt < NT; ++nt) {
      #pragma unroll
      for (int r = 0; r < 8; ++r) V1[nt][r] = DECAY_F * V1[nt][r] + bias1[nt];
    }

    // manually double-buffered K pipeline: 25 steps = prologue + 12x2 + epilogue
    {
      v16h a0, a1, b0[NT], b1[NT];
      a0 = ldA(apk);
      #pragma unroll
      for (int nt = 0; nt < NT; ++nt) b0[nt] = ldB(w1p[nt]);

      #pragma clang loop unroll(disable)
      for (int ks = 0; ks < 24; ks += 2) {
        const int k1 = (ks + 1) * 32;
        const int k2 = (ks + 2) * 32;      // k2 <= 768 < 800: always in range
        a1 = ldA(apk + k1);
        #pragma unroll
        for (int nt = 0; nt < NT; ++nt) b1[nt] = ldB(w1p[nt] + k1);
        #pragma unroll
        for (int nt = 0; nt < NT; ++nt) V1[nt] = wmma16(a0, b0[nt], V1[nt]);
        a0 = ldA(apk + k2);
        #pragma unroll
        for (int nt = 0; nt < NT; ++nt) b0[nt] = ldB(w1p[nt] + k2);
        #pragma unroll
        for (int nt = 0; nt < NT; ++nt) V1[nt] = wmma16(a1, b1[nt], V1[nt]);
      }
      #pragma unroll
      for (int nt = 0; nt < NT; ++nt) V1[nt] = wmma16(a0, b0[nt], V1[nt]);
    }

    // ---- LIF layer 1 (threshold 1.0, hard reset) + stage s1 to LDS ----
    #pragma unroll
    for (int nt = 0; nt < NT; ++nt) {
      #pragma unroll
      for (int r = 0; r < 8; ++r) {
        float v = V1[nt][r];
        float s = (v > 1.0f) ? 1.0f : 0.0f;
        V1[nt][r] = (s > 0.5f) ? 0.0f : v;
        int m = r + 8 * lg;                       // C layout row
        lds_s1[m * H_DIM + wave * 64 + nt * 16 + lx] = (_Float16)s;
      }
    }
    __syncthreads();

    // ---- layer 2 partials: wave w covers K slice [w*64, w*64+64) ----
    v8f c2 = {};
    {
      const int kb = wave * 64;
      v16h a  = ldA(a2p + kb);
      v16h b  = ldB(w2p + kb);
      v16h a2 = ldA(a2p + kb + 32);
      v16h bb = ldB(w2p + kb + 32);
      c2 = wmma16(a, b, c2);
      c2 = wmma16(a2, bb, c2);
    }
    #pragma unroll
    for (int r = 0; r < 8; ++r) lds_red[wave * 256 + r * 32 + lane] = c2[r];
    __syncthreads();

    // ---- wave 0: cross-wave reduce, LIF layer 2, emit spike train ----
    if (wave == 0) {
      #pragma unroll
      for (int r = 0; r < 8; ++r) {
        float s = 0.0f;
        #pragma unroll
        for (int w = 0; w < NWAVE; ++w) s += lds_red[w * 256 + r * 32 + lane];
        float v = DECAY_F * V2[r] + s + bias2;
        float sp = (v > 1.0f) ? 1.0f : 0.0f;
        V2[r] = (sp > 0.5f) ? 0.0f : v;
        cnt[r] += sp;
        if (lx < OUT_DIM) {
          int m = r + 8 * lg;
          trains[((size_t)t * B_TOT + rowBase + m) * OUT_DIM + lx] = sp;
        }
      }
    }
    __syncthreads();   // protect lds_red/lds_spk before next timestep
  }

  if (wave == 0 && lx < OUT_DIM) {
    #pragma unroll
    for (int r = 0; r < 8; ++r) {
      int m = r + 8 * lg;
      counts[(size_t)(rowBase + m) * OUT_DIM + lx] = cnt[r];
    }
  }
}

extern "C" void kernel_launch(void* const* d_in, const int* in_sizes, int n_in,
                              void* d_out, int out_size, void* d_ws, size_t ws_size,
                              hipStream_t stream) {
  const float* input = (const float*)d_in[0];
  const float* W1    = (const float*)d_in[1];
  const float* b1    = (const float*)d_in[2];
  const float* W2    = (const float*)d_in[3];
  const float* b2    = (const float*)d_in[4];

  float* out    = (float*)d_out;
  float* counts = out;                                   // [B,10]
  float* trains = out + (size_t)B_TOT * OUT_DIM;         // [T,B,10]

  _Float16* W1h = (_Float16*)d_ws;                                     // 1024*800*2 B
  _Float16* W2h = (_Float16*)((char*)d_ws + (size_t)H_DIM * KP1 * 2);  // 16*1024*2 B

  conv_w1<<<(H_DIM * KP1 + 255) / 256, 256, 0, stream>>>(W1, W1h);
  conv_w2<<<(OUT_PAD * H_DIM + 255) / 256, 256, 0, stream>>>(W2, W2h);
  snn_fused<<<B_TOT / 16, NWAVE * 32, 0, stream>>>(input, b1, b2, W1h, W2h, counts, trains);
}
